// CausalUncertaintyInference_50783693308201
// MI455X (gfx1250) — compile-verified, matrix operand order checked by reading, not run on hardware
//
#include <hip/hip_runtime.h>
#include <hip/hip_bf16.h>
#include <math.h>

// ---------------- CDNA5 WMMA plumbing (wave32, 16x16x32 f16 -> f32) --------

typedef __attribute__((ext_vector_type(16))) _Float16 v16h;
typedef __attribute__((ext_vector_type(8)))  float    v8f;

union AFrag { v16h v; _Float16 h[16]; };
union CFrag { v8f  v; float     f[8]; };

// A-matrix 16x32 f16 layout (ISA 7.12.2): lanes 0-15 -> K {0..7,16..23},
// lanes 16-31 -> K {8..15,24..31}; element e<8 in VGPR0-3, e>=8 in VGPR4-7.
__device__ __forceinline__ int kmap(int e, int lane) {
  return ((e & 8) ? 16 : 0) + ((lane & 16) ? 8 : 0) + (e & 7);
}

__device__ __forceinline__ v8f wmma16(const AFrag& a, const AFrag& b, const v8f& c) {
  return __builtin_amdgcn_wmma_f32_16x16x32_f16(false, a.v, false, b.v,
                                                (short)0, c, false, false);
}

// Async global->LDS 16B copy (CDNA5 ISA, GV mode) +
// ASYNCcnt fence. Inline asm = portable across ROCm7.2 / amdgpu-toolchain.
__device__ __forceinline__ void async_ld_b128(unsigned lds_dst, const void* gsrc) {
  unsigned long long ga = (unsigned long long)gsrc;
  asm volatile("global_load_async_to_lds_b128 %0, %1, off"
               :: "v"(lds_dst), "v"(ga) : "memory");
}
__device__ __forceinline__ void async_wait0() {
  asm volatile("s_wait_asynccnt 0" ::: "memory");
}

#define NTOK 576   // H*W = 24*24
#define BSZ  2
#define DMODEL 128
#define NHEADS 8
#define HD 16

// ---------------- Kernel 1: f32 -> f16 weight conversion --------------------
__global__ void k_f32_to_f16(const float* __restrict__ src,
                             _Float16* __restrict__ dst, int n) {
  int i = blockIdx.x * 256 + threadIdx.x;
  if (i < n) dst[i] = (_Float16)src[i];
}

// ---------------- Kernel 2: build node features -----------------------------
__global__ void k_build_nf(const float* __restrict__ unc,
                           const float* __restrict__ sw, const float* __restrict__ sb,
                           const float* __restrict__ uw, const float* __restrict__ ub,
                           float* __restrict__ nf, _Float16* __restrict__ nfh) {
  int f = threadIdx.x;             // 0..127
  int p = blockIdx.x % NTOK;
  int b = blockIdx.x / NTOK;
  float val;
  if (f < 64) {
    float gx = (float)(p % 24) * (1.0f / 23.0f);
    float gy = (float)(p / 24) * (1.0f / 23.0f);
    val = gx * sw[f * 2 + 0] + gy * sw[f * 2 + 1] + sb[f];
  } else {
    int g = f - 64;
    float u = unc[b * NTOK + p];
    val = u * uw[g] + ub[g];
  }
  size_t idx = ((size_t)(b * NTOK + p)) * DMODEL + f;
  nf[idx] = val;
  nfh[idx] = (_Float16)val;
}

// ---------------- Kernel 3: templated WMMA GEMM -----------------------------
// C[m,n] = sum_k A[m,k] * W[n, woff+k]  (+bias[n]) (+resid[m,n])
template <bool HAS_BIAS, bool HAS_RES, bool OUT_F32, bool OUT_F16>
__global__ void k_gemm_wmma(const _Float16* __restrict__ A, int lda,
                            const _Float16* __restrict__ W, int ldw, int woff,
                            const float* __restrict__ bias,
                            const float* __restrict__ resid,
                            float* __restrict__ C, _Float16* __restrict__ Ch,
                            int M, int N, int ldc) {
  int wave = (int)((blockIdx.x * blockDim.x + threadIdx.x) >> 5);
  int lane = threadIdx.x & 31;
  int tn = N >> 4;
  int tm = wave / tn;
  int tc = wave % tn;
  if (tm * 16 >= M) return;
  int arow = tm * 16 + (lane & 15);
  int wrow = tc * 16 + (lane & 15);
  CFrag c;
#pragma unroll
  for (int r = 0; r < 8; ++r) c.f[r] = 0.0f;
#pragma unroll
  for (int kb = 0; kb < DMODEL; kb += 32) {
    AFrag a, w;
#pragma unroll
    for (int e = 0; e < 16; ++e) {
      int k = kb + kmap(e, lane);
      a.h[e] = A[(size_t)arow * lda + k];
      w.h[e] = W[(size_t)wrow * ldw + woff + k];
    }
    c.v = wmma16(a, w, c.v);
  }
  float bv = 0.0f;
  int ncol = tc * 16 + (lane & 15);
  if (HAS_BIAS) bv = bias[ncol];
#pragma unroll
  for (int r = 0; r < 8; ++r) {
    int m = tm * 16 + r + ((lane & 16) ? 8 : 0);
    float v = c.f[r] + bv;
    size_t idx = (size_t)m * ldc + ncol;
    if (HAS_RES) v += resid[idx];
    if (OUT_F32) C[idx] = v;
    if (OUT_F16) Ch[idx] = (_Float16)v;
  }
}

// ---------------- Kernel 4: fused attention ---------------------------------
// One wave per (b, head, 16-query block). qkv_h: [b, tok, 384] f16 (q|k|v).
__global__ void k_attn(const _Float16* __restrict__ qkv,
                       _Float16* __restrict__ oh) {
  __shared__ float    sc[16 * NTOK];     // scores (f32)
  __shared__ _Float16 at[16 * NTOK];     // softmax probs (f16)
  int lane = threadIdx.x & 31;
  int qb   = blockIdx.x % 36;
  int head = (blockIdx.x / 36) % NHEADS;
  int b    = blockIdx.x / (36 * NHEADS);
  const _Float16* base = qkv + (size_t)b * NTOK * 384;

  AFrag aq;
  int mrow = qb * 16 + (lane & 15);
#pragma unroll
  for (int e = 0; e < 16; ++e) {
    int k = kmap(e, lane);
    aq.h[e] = (k < HD) ? base[(size_t)mrow * 384 + head * HD + k] : (_Float16)0.0f;
  }
  for (int jb = 0; jb < 36; ++jb) {
    AFrag bk;
    int jcol = jb * 16 + (lane & 15);
#pragma unroll
    for (int e = 0; e < 16; ++e) {
      int k = kmap(e, lane);
      bk.h[e] = (k < HD) ? base[(size_t)jcol * 384 + DMODEL + head * HD + k] : (_Float16)0.0f;
    }
    CFrag c;
#pragma unroll
    for (int r = 0; r < 8; ++r) c.f[r] = 0.0f;
    c.v = wmma16(aq, bk, c.v);
#pragma unroll
    for (int r = 0; r < 8; ++r) {
      int m = r + ((lane & 16) ? 8 : 0);
      sc[m * NTOK + jb * 16 + (lane & 15)] = c.f[r] * 0.25f;  // 1/sqrt(16)
    }
  }
  __syncthreads();
  {
    int rrow = lane & 15;
    int half = lane >> 4;
    float* rowp = sc + rrow * NTOK + half * 288;
    float mx = -1e30f;
    for (int t = 0; t < 288; ++t) mx = fmaxf(mx, rowp[t]);
    mx = fmaxf(mx, __shfl_xor(mx, 16));
    float sum = 0.0f;
    for (int t = 0; t < 288; ++t) { float e = __expf(rowp[t] - mx); rowp[t] = e; sum += e; }
    sum += __shfl_xor(sum, 16);
    float inv = 1.0f / sum;
    _Float16* arow = at + rrow * NTOK + half * 288;
    for (int t = 0; t < 288; ++t) arow[t] = (_Float16)(rowp[t] * inv);
  }
  __syncthreads();
  CFrag c;
#pragma unroll
  for (int r = 0; r < 8; ++r) c.f[r] = 0.0f;
  for (int kb = 0; kb < 18; ++kb) {
    AFrag ap, bv;
#pragma unroll
    for (int e = 0; e < 16; ++e) {
      int j = kb * 32 + kmap(e, lane);
      ap.h[e] = at[(lane & 15) * NTOK + j];
      bv.h[e] = base[(size_t)j * 384 + 2 * DMODEL + head * HD + (lane & 15)];
    }
    c.v = wmma16(ap, bv, c.v);
  }
#pragma unroll
  for (int r = 0; r < 8; ++r) {
    int m = qb * 16 + r + ((lane & 16) ? 8 : 0);
    int d = lane & 15;
    oh[((size_t)(b * NTOK + m)) * DMODEL + head * HD + d] = (_Float16)c.f[r];
  }
}

// ---------------- Kernel 5: fused pairwise causal-strength ------------------
// Workgroup = (b, j-tile of 16, i-chunk of 72). The bj tile (8KB) and the
// ENTIRE w2 weight matrix (16KB) are staged into LDS once via async
// global->LDS b128 copies, then reused across 72 i iterations: per i, the
// relu(ai+bj+b1) tile is built in LDS (f16, padded rows), and 4 waves run
// the K=128 WMMA loop entirely out of LDS. This cuts L2 traffic ~25x vs a
// per-(b,i,jtile) grid (w2/bj no longer refetched per tile).
#define PAIR_LDK (DMODEL + 8)   // +8 halves pad -> 4-dword bank skew per row
#define W2_LDK   (DMODEL + 8)
#define ICHUNK   72
__global__ void k_pair(const float* __restrict__ ai, const float* __restrict__ bj,
                       const float* __restrict__ b1, const _Float16* __restrict__ w2,
                       const float* __restrict__ b2, const float* __restrict__ w3,
                       const float* __restrict__ b3, float* __restrict__ out) {
  __shared__ _Float16 w2s[64 * W2_LDK];        // 17408 B
  __shared__ float    bjt[16 * DMODEL];        // 8192 B
  __shared__ float    arow[DMODEL];            // 512 B
  __shared__ _Float16 ptile[16 * PAIR_LDK];    // 4352 B
  __shared__ float    part[16 * 4];

  int jt = blockIdx.x % 36;
  int ic = (blockIdx.x / 36) % (NTOK / ICHUNK);
  int b  = blockIdx.x / (36 * (NTOK / ICHUNK));
  int tid = threadIdx.x;                        // 128 threads / 4 waves
  int wave = tid >> 5, lane = tid & 31;

  // ---- async-stage w2 (64 rows x 128 halves, padded LDS rows) ----
  {
    unsigned lbase = (unsigned)(unsigned long long)(void*)w2s;
#pragma unroll
    for (int it = 0; it < 8; ++it) {
      int t = tid + it * 128;                  // 1024 chunks of 16B
      int row = t >> 4, ch = t & 15;
      async_ld_b128(lbase + (unsigned)(row * W2_LDK * 2 + ch * 16),
                    w2 + (size_t)row * DMODEL + ch * 8);
    }
  }
  // ---- async-stage bj tile (16 rows x 128 f32, contiguous) ----
  {
    unsigned lbase = (unsigned)(unsigned long long)(void*)bjt;
    const float* src0 = bj + ((size_t)(b * NTOK + jt * 16)) * DMODEL;
#pragma unroll
    for (int it = 0; it < 4; ++it) {
      int t = tid + it * 128;                  // 512 chunks of 16B
      async_ld_b128(lbase + (unsigned)(t * 16), src0 + t * 4);
    }
  }
  async_wait0();
  __syncthreads();

  int fcol = wave * 16 + (lane & 15);          // h2 channel (0..63)
  int jloc = lane & 15;
  int kofs = (lane & 16) ? 8 : 0;
  float bias2 = b2[fcol];
  float w3v   = w3[fcol];
  float b3v   = b3[0];

  int i0 = ic * ICHUNK;
  for (int ii = 0; ii < ICHUNK; ++ii) {
    int i = i0 + ii;
    arow[tid] = ai[((size_t)(b * NTOK + i)) * DMODEL + tid] + b1[tid];
    __syncthreads();

    // cooperative pair-tile build: 16 rows x 128 k, 8 threads per row
    {
      int j  = tid >> 3;
      int k0 = (tid & 7) * 16;
      const float* bjr = &bjt[j * DMODEL];
#pragma unroll
      for (int t = 0; t < 16; ++t) {
        int k = k0 + t;
        ptile[j * PAIR_LDK + k] = (_Float16)fmaxf(arow[k] + bjr[k], 0.0f);
      }
    }
    __syncthreads();

    CFrag c;
#pragma unroll
    for (int r = 0; r < 8; ++r) c.f[r] = 0.0f;
#pragma unroll
    for (int kb = 0; kb < DMODEL; kb += 32) {
      AFrag a, w;
      const _Float16* prow = &ptile[jloc * PAIR_LDK + kb + kofs];
      const _Float16* wrow = &w2s[fcol * W2_LDK + kb + kofs];
#pragma unroll
      for (int e = 0; e < 8; ++e) {            // two contiguous 16B chunks each
        a.h[e]     = prow[e];
        a.h[e + 8] = prow[16 + e];
        w.h[e]     = wrow[e];
        w.h[e + 8] = wrow[16 + e];
      }
      c.v = wmma16(a, w, c.v);
    }
    float pr[8];
#pragma unroll
    for (int r = 0; r < 8; ++r)
      pr[r] = fmaxf(c.f[r] + bias2, 0.0f) * w3v;
#pragma unroll
    for (int r = 0; r < 8; ++r) {
      pr[r] += __shfl_xor(pr[r], 1);
      pr[r] += __shfl_xor(pr[r], 2);
      pr[r] += __shfl_xor(pr[r], 4);
      pr[r] += __shfl_xor(pr[r], 8);
    }
    if ((lane & 15) == 0) {
      int mbase = (lane & 16) ? 8 : 0;
#pragma unroll
      for (int r = 0; r < 8; ++r) part[(mbase + r) * 4 + wave] = pr[r];
    }
    __syncthreads();
    if (tid < 16) {
      float t = part[tid * 4] + part[tid * 4 + 1] + part[tid * 4 + 2] +
                part[tid * 4 + 3] + b3v;
      float s = 1.0f / (1.0f + __expf(-t));
      int j = jt * 16 + tid;
      if (j == i) s = 0.0f;
      out[((size_t)(b * NTOK + i)) * NTOK + j] = s;
    }
  }
}

// ---------------- host launch ----------------------------------------------
extern "C" void kernel_launch(void* const* d_in, const int* in_sizes, int n_in,
                              void* d_out, int out_size, void* d_ws, size_t ws_size,
                              hipStream_t stream) {
  const float* unc        = (const float*)d_in[0];
  const float* spatial_w  = (const float*)d_in[1];
  const float* spatial_b  = (const float*)d_in[2];
  const float* unc_w      = (const float*)d_in[3];
  const float* unc_b      = (const float*)d_in[4];
  const float* in_proj_w  = (const float*)d_in[5];   // (3, 384, 128)
  const float* in_proj_b  = (const float*)d_in[6];   // (3, 384)
  const float* out_proj_w = (const float*)d_in[7];   // (3, 128, 128)
  const float* out_proj_b = (const float*)d_in[8];   // (3, 128)
  const float* cs_w1      = (const float*)d_in[9];   // (128, 256)
  const float* cs_b1      = (const float*)d_in[10];  // (128,)
  const float* cs_w2      = (const float*)d_in[11];  // (64, 128)
  const float* cs_b2      = (const float*)d_in[12];  // (64,)
  const float* cs_w3      = (const float*)d_in[13];  // (1, 64)
  const float* cs_b3      = (const float*)d_in[14];  // (1,)
  float* out = (float*)d_out;

  char* ws = (char*)d_ws;
  size_t off = 0;
  auto carve = [&](size_t bytes) -> void* {
    void* p = ws + off;
    off = (off + bytes + 255) & ~(size_t)255;
    return p;
  };
  const int MT = BSZ * NTOK;  // 1152 tokens total
  float*    nf    = (float*)   carve((size_t)MT * DMODEL * 4);
  _Float16* nfh   = (_Float16*)carve((size_t)MT * DMODEL * 2);
  _Float16* qkvh  = (_Float16*)carve((size_t)MT * 3 * DMODEL * 2);
  _Float16* oh    = (_Float16*)carve((size_t)MT * DMODEL * 2);
  float*    ai    = (float*)   carve((size_t)MT * DMODEL * 4);
  float*    bj    = (float*)   carve((size_t)MT * DMODEL * 4);
  _Float16* inWh  = (_Float16*)carve((size_t)3 * 384 * DMODEL * 2);
  _Float16* outWh = (_Float16*)carve((size_t)3 * DMODEL * DMODEL * 2);
  _Float16* w1h   = (_Float16*)carve((size_t)DMODEL * 256 * 2);
  _Float16* w2h   = (_Float16*)carve((size_t)64 * DMODEL * 2);
  (void)ws_size; (void)n_in; (void)in_sizes; (void)out_size;

  // ---- 1) weights to f16 ----
  {
    int n1 = 3 * 384 * DMODEL;
    k_f32_to_f16<<<(n1 + 255) / 256, 256, 0, stream>>>(in_proj_w, inWh, n1);
    int n2 = 3 * DMODEL * DMODEL;
    k_f32_to_f16<<<(n2 + 255) / 256, 256, 0, stream>>>(out_proj_w, outWh, n2);
    int n3 = DMODEL * 256;
    k_f32_to_f16<<<(n3 + 255) / 256, 256, 0, stream>>>(cs_w1, w1h, n3);
    int n4 = 64 * DMODEL;
    k_f32_to_f16<<<(n4 + 255) / 256, 256, 0, stream>>>(cs_w2, w2h, n4);
  }

  // ---- 2) node features ----
  k_build_nf<<<BSZ * NTOK, DMODEL, 0, stream>>>(unc, spatial_w, spatial_b,
                                                unc_w, unc_b, nf, nfh);

  // ---- 3) transformer layers ----
  for (int l = 0; l < 3; ++l) {
    {
      int tiles = (MT / 16) * (384 / 16);            // 1728
      k_gemm_wmma<true, false, false, true><<<tiles / 4, 128, 0, stream>>>(
          nfh, DMODEL, inWh + (size_t)l * 384 * DMODEL, DMODEL, 0,
          in_proj_b + (size_t)l * 384, nullptr,
          nullptr, qkvh, MT, 384, 384);
    }
    k_attn<<<BSZ * NHEADS * 36, 32, 0, stream>>>(qkvh, oh);
    {
      int tiles = (MT / 16) * (DMODEL / 16);         // 576
      k_gemm_wmma<true, true, true, true><<<tiles / 4, 128, 0, stream>>>(
          oh, DMODEL, outWh + (size_t)l * DMODEL * DMODEL, DMODEL, 0,
          out_proj_b + (size_t)l * DMODEL, nf,
          nf, nfh, MT, DMODEL, DMODEL);
    }
  }

  // ---- 4) ai = nf @ w1a^T, bj = nf @ w1b^T ----
  {
    int tiles = (MT / 16) * (DMODEL / 16);           // 576
    k_gemm_wmma<false, false, true, false><<<tiles / 4, 128, 0, stream>>>(
        nfh, DMODEL, w1h, 256, 0, nullptr, nullptr, ai, nullptr, MT, DMODEL, DMODEL);
    k_gemm_wmma<false, false, true, false><<<tiles / 4, 128, 0, stream>>>(
        nfh, DMODEL, w1h, 256, DMODEL, nullptr, nullptr, bj, nullptr, MT, DMODEL, DMODEL);
  }

  // ---- 5) pairwise causal strength (dominant stage) ----
  // grid: b x i-chunks x j-tiles = 2 * 8 * 36 = 576 blocks
  k_pair<<<BSZ * (NTOK / ICHUNK) * 36, 128, 0, stream>>>(
      ai, bj, cs_b1, w2h, cs_b2, cs_w3, cs_b3, out);
}
